// GroupQueryAttention_84361747628064
// MI455X (gfx1250) — matrix operand order repaired and checked by reading, not compiled, not run on hardware
//
#include <hip/hip_runtime.h>
#include <hip/hip_bf16.h>

// ---------------------------------------------------------------------------
// GQA for MI455X (gfx1250, wave32). All GEMMs via v_wmma_f32_16x16x32_f16.
// Fragment layouts used here (contiguous k for A and B alike, not the ISA table order):
//   A 16x32 f16 : lane L holds row M=L%16; halves e=0..15 map to
//                 K = e + 16*(L/16)  -> one contiguous 16-half chunk (read as two 8-half loads)
//   B 32x16 f16 : lane L holds col N=L%16; halves e map to K = e + 16*(L/16)
//                 -> one contiguous 16-half chunk of the (column-major) B, i.e.
//                 a contiguous row of W when computing x @ W^T
//   C/D 16x16 f32: VGPR r, lane L -> M = r + 8*(L/16), N = L%16
// ---------------------------------------------------------------------------

typedef __attribute__((ext_vector_type(16))) _Float16 v16h;
typedef __attribute__((ext_vector_type(8)))  _Float16 v8h;
typedef __attribute__((ext_vector_type(4)))  _Float16 v4h;
typedef __attribute__((ext_vector_type(8)))  float    v8f;
typedef __attribute__((ext_vector_type(4)))  float    v4f;

#define DEV __device__ __forceinline__

static constexpr int B_ = 2, S_ = 2048, E_ = 768, H_ = 12, D_ = 64, G_ = 4, T_ = 512;

DEV v16h load_a_frag(const _Float16* __restrict__ A, int lda) {
  const int lane = threadIdx.x & 31;
  const int m = lane & 15, hi = lane >> 4;
  const _Float16* p = A + (size_t)m * lda + 16 * hi;
  v8h lo = *(const v8h*)p;           // K = 16*hi .. 16*hi+7
  v8h hh = *(const v8h*)(p + 8);     // K = 16*hi+8 .. 16*hi+15
  v16h r;
#pragma unroll
  for (int i = 0; i < 8; ++i) { r[i] = lo[i]; r[8 + i] = hh[i]; }
  return r;
}

// B[k][n] = W[n][k] (row of W is a column of B). ldw = row stride of W.
DEV v16h load_b_frag(const _Float16* __restrict__ W, int ldw) {
  const int lane = threadIdx.x & 31;
  const int n = lane & 15, hi = lane >> 4;
  return *(const v16h*)(W + (size_t)n * ldw + 16 * hi);
}

DEV v8f wmma_f16(v16h a, v16h b, v8f c) {
  return __builtin_amdgcn_wmma_f32_16x16x32_f16(false, a, false, b, (short)0, c,
                                                false, false);
}

// xor-tree max over the 16-lane half (rows live per-half in the C layout).
// DPP: after quad xor1/xor2 steps, quads are uniform, so half/full row
// mirrors are equivalent to xor4/xor8. 4 VALU ops, no LDS traffic.
#if __has_builtin(__builtin_amdgcn_update_dpp)
template <int CTRL>
DEV float dpp_max_step(float v) {
  int x = __builtin_bit_cast(int, v);
  int y = __builtin_amdgcn_update_dpp(x, x, CTRL, 0xF, 0xF, true);
  return fmaxf(v, __builtin_bit_cast(float, y));
}
DEV float rowmax16(float v) {
  v = dpp_max_step<0xB1>(v);   // quad_perm(1,0,3,2)  : xor 1
  v = dpp_max_step<0x4E>(v);   // quad_perm(2,3,0,1)  : xor 2
  v = dpp_max_step<0x141>(v);  // row_half_mirror     : xor 4 (quads uniform)
  v = dpp_max_step<0x140>(v);  // row_mirror          : xor 8 (octs uniform)
  return v;
}
#else
DEV float rowmax16(float v) {
  for (int msk = 1; msk < 16; msk <<= 1) v = fmaxf(v, __shfl_xor(v, msk, 32));
  return v;
}
#endif

// ---------------------------------------------------------------------------
// fp32 -> fp16 conversion (4 elems / thread)
// ---------------------------------------------------------------------------
__global__ void cvt_f32_f16(const float* __restrict__ in, _Float16* __restrict__ out,
                            int n4) {
  int i = blockIdx.x * blockDim.x + threadIdx.x;
  if (i < n4) {
    v4f x = *(const v4f*)(in + 4 * (size_t)i);
    v4h y;
#pragma unroll
    for (int k = 0; k < 4; ++k) y[k] = (_Float16)x[k];
    *(v4h*)(out + 4 * (size_t)i) = y;
  }
}

// ---------------------------------------------------------------------------
// Generic wave-tile GEMM:  C[M,N] = A[M,K] @ W[N,K]^T + bias[N]
// One wave (blockDim=32) per 32x64 output tile (2 A-frags x 4 B-frags -> 8
// WMMAs per k-step, each B fragment reused twice). Batched over blockIdx.z
// with z = b*gdim + g and independent (b,g) base strides for A/W/bias/C.
// OMODE: 0 = f16 C, 1 = f16 C stored transposed (C[n][m], ldc = col stride),
//        2 = f32 C.
// ---------------------------------------------------------------------------
template <int OMODE>
__global__ void gemm16(const _Float16* __restrict__ A, long sAb, long sAg, int lda,
                       const _Float16* __restrict__ W, long sWg, int ldw,
                       const float* __restrict__ bias, long sBg,
                       void* __restrict__ Cv, long sCb, long sCg, int ldc,
                       int M, int N, int K, int gdim) {
  const int z = blockIdx.z;
  const int b = z / gdim, g = z % gdim;
  A    += (size_t)b * sAb + (size_t)g * sAg;
  W    += (size_t)g * sWg;
  bias += (size_t)g * sBg;
  const int n0 = blockIdx.x * 64;
  const int m0 = blockIdx.y * 32;
  const int lane = threadIdx.x & 31;
  const int n = lane & 15, hi = lane >> 4;

  v8f acc[2][4] = {};
  for (int k0 = 0; k0 < K; k0 += 32) {
    v16h a0 = load_a_frag(A + (size_t)m0 * lda + k0, lda);
    v16h a1 = load_a_frag(A + (size_t)(m0 + 16) * lda + k0, lda);
#pragma unroll
    for (int j = 0; j < 4; ++j) {
      v16h bf = load_b_frag(W + (size_t)(n0 + 16 * j) * ldw + k0, ldw);
      acc[0][j] = wmma_f16(a0, bf, acc[0][j]);
      acc[1][j] = wmma_f16(a1, bf, acc[1][j]);
    }
  }

#pragma unroll
  for (int i = 0; i < 2; ++i) {
    const int mb = m0 + 16 * i;
    if (OMODE == 0) {
      _Float16* C = (_Float16*)Cv + (size_t)b * sCb + (size_t)g * sCg;
#pragma unroll
      for (int j = 0; j < 4; ++j) {
        float bb = bias[n0 + 16 * j + n];
#pragma unroll
        for (int r = 0; r < 8; ++r)
          C[(size_t)(mb + r + 8 * hi) * ldc + n0 + 16 * j + n] =
              (_Float16)(acc[i][j][r] + bb);
      }
    } else if (OMODE == 1) {
      _Float16* C = (_Float16*)Cv + (size_t)b * sCb + (size_t)g * sCg;
#pragma unroll
      for (int j = 0; j < 4; ++j) {
        float bb = bias[n0 + 16 * j + n];
        v8h pk;
#pragma unroll
        for (int r = 0; r < 8; ++r) pk[r] = (_Float16)(acc[i][j][r] + bb);
        // rows M = mb+8*hi..+7 are contiguous -> one 16B store per lane
        *(v8h*)(C + (size_t)(n0 + 16 * j + n) * ldc + mb + 8 * hi) = pk;
      }
    } else {
      float* C = (float*)Cv + (size_t)b * sCb + (size_t)g * sCg;
#pragma unroll
      for (int j = 0; j < 4; ++j) {
        float bb = bias[n0 + 16 * j + n];
#pragma unroll
        for (int r = 0; r < 8; ++r)
          C[(size_t)(mb + r + 8 * hi) * ldc + n0 + 16 * j + n] =
              acc[i][j][r] + bb;
      }
    }
  }
}

// ---------------------------------------------------------------------------
// Flash attention. One wave handles (b,g,h, 16-row Q tile); streams over S in
// 32-column chunks with online (log2-domain) softmax. The softmax scale
// (1/sqrt(D) * log2e) is folded into the Q fragments once, so WMMA emits
// scores directly in the log2 domain. Row max via DPP xor tree; row sum via
// an extra WMMA against an all-ones B fragment (exact). P is re-laid-out
// through a per-wave 1KB LDS buffer to form the A-fragment for the P·V WMMAs.
//   qi : [B*G, T, E] f16      ki : [B*G, S, E] f16
//   vt : [B*G, E, S] f16 (transposed V)      ctx: [B*G, T, E] f16
// ---------------------------------------------------------------------------
__global__ void flash_attn(const _Float16* __restrict__ qi,
                           const _Float16* __restrict__ ki,
                           const _Float16* __restrict__ vt,
                           _Float16* __restrict__ ctx) {
  __shared__ __align__(16) _Float16 pbuf[4][16][32];
  const int wave = threadIdx.x >> 5;
  const int lane = threadIdx.x & 31;
  const int unit = blockIdx.x * 4 + wave;      // B*G*H*(T/16) = 3072 units
  const int tt = unit & 31;                    // T/16 = 32
  int tmp = unit >> 5;
  const int h = tmp % H_; tmp /= H_;
  const int g = tmp % G_;
  const int b = tmp / G_;
  const int bg = b * G_ + g;
  const int t0 = tt * 16;

  const _Float16* qbase = qi + ((size_t)bg * T_ + t0) * E_ + h * D_;
  const _Float16* kbase = ki + (size_t)bg * S_ * E_ + h * D_;
  const _Float16* vbase = vt + (size_t)bg * (size_t)E_ * S_ + (size_t)h * D_ * S_;

  // fold (1/sqrt(D)) * log2(e) into Q once: scores come out in log2 domain
  const _Float16 qs = (_Float16)(0.125f * 1.44269504088896f);
  v16h qa0 = load_a_frag(qbase, E_);
  v16h qa1 = load_a_frag(qbase + 32, E_);
#pragma unroll
  for (int i = 0; i < 16; ++i) { qa0[i] *= qs; qa1[i] *= qs; }

  v16h ones;
#pragma unroll
  for (int i = 0; i < 16; ++i) ones[i] = (_Float16)1.0f;

  const int n = lane & 15, hi = lane >> 4;
  float mrow[8];
  v8f o[4] = {};
  v8f lsum = {};
#pragma unroll
  for (int r = 0; r < 8; ++r) mrow[r] = -1e30f;

  for (int s0 = 0; s0 < S_; s0 += 32) {
    v8f sc[2] = {};
#pragma unroll
    for (int c = 0; c < 2; ++c) {
      const _Float16* kp = kbase + (size_t)(s0 + 16 * c) * E_;
      v16h b0 = load_b_frag(kp, E_);
      v16h b1 = load_b_frag(kp + 32, E_);
      sc[c] = wmma_f16(qa0, b0, sc[c]);
      sc[c] = wmma_f16(qa1, b1, sc[c]);
    }

    float alpha[8];
#pragma unroll
    for (int r = 0; r < 8; ++r) {
      float mx = rowmax16(fmaxf(sc[0][r], sc[1][r]));
      float mnew = fmaxf(mrow[r], mx);
      alpha[r] = exp2f(mrow[r] - mnew);
      mrow[r] = mnew;
    }
#pragma unroll
    for (int r = 0; r < 8; ++r) {
      sc[0][r] = exp2f(sc[0][r] - mrow[r]);
      sc[1][r] = exp2f(sc[1][r] - mrow[r]);
    }

    // stage P (C-layout regs) -> LDS row-major [16][32] -> reload as A-frag
#pragma unroll
    for (int c = 0; c < 2; ++c)
#pragma unroll
      for (int r = 0; r < 8; ++r)
        pbuf[wave][r + 8 * hi][16 * c + n] = (_Float16)sc[c][r];
    asm volatile("s_wait_dscnt 0" ::: "memory");   // CDNA5 split DS counter
    v16h pa = load_a_frag(&pbuf[wave][0][0], 32);

    // rescale running accumulators, then accumulate this chunk
#pragma unroll
    for (int r = 0; r < 8; ++r) lsum[r] *= alpha[r];
    lsum = wmma_f16(pa, ones, lsum);               // rowsum(P) per row, any col
#pragma unroll
    for (int j = 0; j < 4; ++j) {
#pragma unroll
      for (int r = 0; r < 8; ++r) o[j][r] *= alpha[r];
      v16h vb = load_b_frag(vbase + (size_t)(16 * j) * S_ + s0, S_);
      o[j] = wmma_f16(pa, vb, o[j]);
    }
  }

  _Float16* cb = ctx + ((size_t)bg * T_ + t0) * E_ + h * D_;
#pragma unroll
  for (int j = 0; j < 4; ++j)
#pragma unroll
    for (int r = 0; r < 8; ++r)
      cb[(size_t)(r + 8 * hi) * E_ + 16 * j + n] = (_Float16)(o[j][r] / lsum[r]);
}

// ---------------------------------------------------------------------------
extern "C" void kernel_launch(void* const* d_in, const int* in_sizes, int n_in,
                              void* d_out, int out_size, void* d_ws, size_t ws_size,
                              hipStream_t stream) {
  (void)in_sizes; (void)n_in; (void)out_size; (void)ws_size;
  const float* query = (const float*)d_in[0];
  const float* key   = (const float*)d_in[1];
  const float* value = (const float*)d_in[2];
  const float* Wqg   = (const float*)d_in[3];
  const float* bqg   = (const float*)d_in[4];
  const float* Wk    = (const float*)d_in[5];
  const float* bk    = (const float*)d_in[6];
  const float* Wv    = (const float*)d_in[7];
  const float* bv    = (const float*)d_in[8];
  const float* Wq_in = (const float*)d_in[9];
  const float* bq_in = (const float*)d_in[10];
  const float* Wk_in = (const float*)d_in[11];
  const float* bk_in = (const float*)d_in[12];
  const float* Wv_in = (const float*)d_in[13];
  const float* bv_in = (const float*)d_in[14];
  const float* Wout  = (const float*)d_in[15];
  const float* bout  = (const float*)d_in[16];

  const size_t BSE = (size_t)B_ * S_ * E_;   // 3,145,728
  const size_t EE  = (size_t)E_ * E_;        // 589,824
  const size_t GEE = (size_t)G_ * E_ * E_;   // 2,359,296

  _Float16* w = (_Float16*)d_ws;
  _Float16* query16 = w; w += BSE;
  _Float16* key16   = w; w += BSE;
  _Float16* value16 = w; w += BSE;
  _Float16* q16     = w; w += BSE;
  _Float16* k16     = w; w += BSE;
  _Float16* v16     = w; w += BSE;
  _Float16* Wqg16   = w; w += EE;
  _Float16* Wk16    = w; w += EE;
  _Float16* Wv16    = w; w += EE;
  _Float16* Wqin16  = w; w += GEE;
  _Float16* Wkin16  = w; w += GEE;
  _Float16* Wvin16  = w; w += GEE;
  _Float16* Wout16  = w; w += GEE;
  _Float16* ki16    = w; w += 4 * BSE;       // [B*G, S, E]
  _Float16* vt16    = w; w += 4 * BSE;       // [B*G, E, S]
  _Float16* qi16  = query16;   // dead after G1, reused  [B*G, T, E]
  _Float16* ctx16 = key16;     // dead after G2, reused  [B*G, T, E]

  auto cvt = [&](const float* src, _Float16* dst, size_t nelem) {
    int n4 = (int)(nelem / 4);
    cvt_f32_f16<<<(n4 + 255) / 256, 256, 0, stream>>>(src, dst, n4);
  };
  cvt(query, query16, BSE);
  cvt(key,   key16,   BSE);
  cvt(value, value16, BSE);
  cvt(Wqg,   Wqg16,   EE);
  cvt(Wk,    Wk16,    EE);
  cvt(Wv,    Wv16,    EE);
  cvt(Wq_in, Wqin16,  GEE);
  cvt(Wk_in, Wkin16,  GEE);
  cvt(Wv_in, Wvin16,  GEE);
  cvt(Wout,  Wout16,  GEE);

  const dim3 blk(32);

  // G1-G3: outer projections  [B*S,E] @ [E,E]^T
  gemm16<0><<<dim3(E_ / 64, (B_ * S_) / 32, 1), blk, 0, stream>>>(
      query16, 0, 0, E_, Wqg16, 0, E_, bqg, 0, q16, 0, 0, E_,
      B_ * S_, E_, E_, 1);
  gemm16<0><<<dim3(E_ / 64, (B_ * S_) / 32, 1), blk, 0, stream>>>(
      key16, 0, 0, E_, Wk16, 0, E_, bk, 0, k16, 0, 0, E_,
      B_ * S_, E_, E_, 1);
  gemm16<0><<<dim3(E_ / 64, (B_ * S_) / 32, 1), blk, 0, stream>>>(
      value16, 0, 0, E_, Wv16, 0, E_, bv, 0, v16, 0, 0, E_,
      B_ * S_, E_, E_, 1);

  // G4: qi[b,g,t,:] = q[b, t*G+g, :] @ Wq_in[g]^T  (grouping folded into lda)
  gemm16<0><<<dim3(E_ / 64, T_ / 32, B_ * G_), blk, 0, stream>>>(
      q16, (long)(S_ * E_), (long)E_, G_ * E_,
      Wqin16, (long)EE, E_, bq_in, (long)E_,
      qi16, (long)(G_ * T_ * E_), (long)(T_ * E_), E_,
      T_, E_, E_, G_);

  // G5: ki[b,g,s,:] = k[b,s,:] @ Wk_in[g]^T
  gemm16<0><<<dim3(E_ / 64, S_ / 32, B_ * G_), blk, 0, stream>>>(
      k16, (long)(S_ * E_), 0L, E_,
      Wkin16, (long)EE, E_, bk_in, (long)E_,
      ki16, (long)((size_t)G_ * S_ * E_), (long)((size_t)S_ * E_), E_,
      S_, E_, E_, G_);

  // G6: vt[b,g,f,s] = (v[b,s,:] @ Wv_in[g]^T)^T  (transposed store)
  gemm16<1><<<dim3(E_ / 64, S_ / 32, B_ * G_), blk, 0, stream>>>(
      v16, (long)(S_ * E_), 0L, E_,
      Wvin16, (long)EE, E_, bv_in, (long)E_,
      vt16, (long)((size_t)G_ * E_ * S_), (long)((size_t)E_ * S_), S_,
      S_, E_, E_, G_);

  // G7: flash attention -> ctx16 [B*G, T, E]
  flash_attn<<<dim3((B_ * G_ * H_ * (T_ / 16)) / 4), dim3(128), 0, stream>>>(
      qi16, ki16, vt16, ctx16);

  // G8: out[b, t*G+g, :] = ctx[b,g,t,:] @ Wout[g]^T  (f32, interleave in ldc)
  gemm16<2><<<dim3(E_ / 64, T_ / 32, B_ * G_), blk, 0, stream>>>(
      ctx16, (long)(G_ * T_ * E_), (long)(T_ * E_), E_,
      Wout16, (long)EE, E_, bout, (long)E_,
      d_out, (long)(S_ * E_), (long)E_, G_ * E_,
      T_, E_, E_, G_);
}
